// Layer_1571958030581
// MI455X (gfx1250) — matrix operand order repaired and checked by reading, not compiled, not run on hardware
//
#include <hip/hip_runtime.h>

#define NN 50000
#define IN_D 512
#define HD 512
#define NH 8
#define DH 64
#define SLOPE 0.2f

typedef __attribute__((ext_vector_type(16))) _Float16 v16h;
typedef __attribute__((ext_vector_type(8)))  _Float16 v8h;
typedef __attribute__((ext_vector_type(8)))  float    v8f;

__device__ __forceinline__ float lrelu(float x) { return x > 0.f ? x : SLOPE * x; }

// order-preserving float<->uint encoding for atomicMax-based segment max
__device__ __forceinline__ unsigned encf(float x) {
  unsigned b = __float_as_uint(x);
  return (b & 0x80000000u) ? ~b : (b | 0x80000000u);
}
__device__ __forceinline__ float decf(unsigned u) {
  return (u & 0x80000000u) ? __uint_as_float(u & 0x7FFFFFFFu) : __uint_as_float(~u);
}

// ---------------------------------------------------------------------------
// Kernel 0: zero-init output, segment-sum and encoded segment-max buffers
// ---------------------------------------------------------------------------
__global__ void init_kernel(float* __restrict__ out, float* __restrict__ ssum,
                            unsigned* __restrict__ mEnc) {
  size_t stride = (size_t)gridDim.x * blockDim.x;
  size_t i0 = (size_t)blockIdx.x * blockDim.x + threadIdx.x;
  for (size_t j = i0; j < (size_t)NN * HD; j += stride) out[j] = 0.f;
  for (size_t j = i0; j < (size_t)NN * NH; j += stride) { ssum[j] = 0.f; mEnc[j] = 0u; }
}

// ---------------------------------------------------------------------------
// Kernel P: one-time f32 -> f16 conversion of feat and W (removes all
// v_cvt work from the WMMA hot loop, halves GEMM streaming bytes)
// ---------------------------------------------------------------------------
__global__ void cvt_f16_kernel(const float* __restrict__ feat, const float* __restrict__ W,
                               _Float16* __restrict__ featH, _Float16* __restrict__ WH) {
  size_t stride = (size_t)gridDim.x * blockDim.x;
  size_t i0 = (size_t)blockIdx.x * blockDim.x + threadIdx.x;
  for (size_t j = i0; j < (size_t)NN * IN_D; j += stride) featH[j] = (_Float16)feat[j];
  for (size_t j = i0; j < (size_t)HD * IN_D; j += stride) WH[j] = (_Float16)W[j];
}

// ---------------------------------------------------------------------------
// Kernel 1: h = leaky_relu(featH @ WH^T + b) via WMMA f32_16x16x32_f16
// One wave per 16x128 output tile: 8 accumulators, A loaded once per K-step.
// blockIdx.x: row tile (M/16), blockIdx.y: col tile (N/128)
// ---------------------------------------------------------------------------
__global__ __launch_bounds__(32) void gemm_lrelu_kernel(
    const _Float16* __restrict__ featH, const _Float16* __restrict__ WH,
    const float* __restrict__ bias, float* __restrict__ h)
{
  const int lane = threadIdx.x;
  const int r    = lane & 15;
  const int half = lane >> 4;
  const int m0 = blockIdx.x * 16;
  const int n0 = blockIdx.y * 128;

  // A (16x32 f16): lanes 0-15 row M=lane K=0..15; lanes 16-31 K=16..31 (contiguous)
  const _Float16* arow = featH + (size_t)(m0 + r) * IN_D + half * 16;
  // B (32x16 f16): lanes 0-15 col N=lane K=0..15; lanes 16-31 K=16..31 (contiguous)
  const _Float16* brow = WH + (size_t)(n0 + r) * IN_D + half * 16;

  v8f acc[8] = {};

  for (int k0 = 0; k0 < IN_D; k0 += 32) {
    v8h alo = *(const v8h*)(arow + k0);
    v8h ahi = *(const v8h*)(arow + k0 + 8);
    v16h a = __builtin_shufflevector(alo, ahi,
                                     0, 1, 2, 3, 4, 5, 6, 7,
                                     8, 9, 10, 11, 12, 13, 14, 15);
#pragma unroll
    for (int t = 0; t < 8; ++t) {
      v16h b = *(const v16h*)(brow + (size_t)t * 16 * IN_D + k0);
      acc[t] = __builtin_amdgcn_wmma_f32_16x16x32_f16(
          false, a, false, b, (short)0, acc[t], false, false);
    }
  }

  // C/D layout: VGPR i -> M = m0 + i + 8*half, N = n0 + t*16 + r
#pragma unroll
  for (int t = 0; t < 8; ++t) {
    int n = n0 + t * 16 + r;
    float bb = bias[n];
#pragma unroll
    for (int i = 0; i < 8; ++i) {
      int m = m0 + i + half * 8;
      h[(size_t)m * HD + n] = lrelu(acc[t][i] + bb);
    }
  }
}

// ---------------------------------------------------------------------------
// Kernel 2: per-(node, head) attention logits e_src/e_dst
// ---------------------------------------------------------------------------
__global__ __launch_bounds__(256) void logits_kernel(
    const float* __restrict__ h, const float* __restrict__ was,
    const float* __restrict__ wad, float* __restrict__ es, float* __restrict__ ed)
{
  int gid = blockIdx.x * blockDim.x + threadIdx.x;
  if (gid >= NN * NH) return;
  int node = gid >> 3, hh = gid & 7;
  const float* hp = h + (size_t)node * HD + hh * DH;
  const float* sp = was + hh * DH;
  const float* dp = wad + hh * DH;
  float as = 0.f, ad = 0.f;
#pragma unroll
  for (int k = 0; k < DH; k += 4) {
    float4 hv = *(const float4*)(hp + k);
    float4 sv = *(const float4*)(sp + k);
    float4 dv = *(const float4*)(dp + k);
    as += hv.x * sv.x + hv.y * sv.y + hv.z * sv.z + hv.w * sv.w;
    ad += hv.x * dv.x + hv.y * dv.y + hv.z * dv.z + hv.w * dv.w;
  }
  es[gid] = as;
  ed[gid] = ad;
}

// ---------------------------------------------------------------------------
// Kernel 3: edge logits + encoded atomicMax segment-max over dst
// ---------------------------------------------------------------------------
__global__ __launch_bounds__(256) void edge_max_kernel(
    const int* __restrict__ src, const int* __restrict__ dst,
    const float* __restrict__ es, const float* __restrict__ ed,
    unsigned* __restrict__ mEnc, int E)
{
  int e = blockIdx.x * blockDim.x + threadIdx.x;
  if (e >= E) return;
  int s = src[e], d = dst[e];
#pragma unroll
  for (int hh = 0; hh < NH; ++hh) {
    float x = lrelu(es[(size_t)s * NH + hh] + ed[(size_t)d * NH + hh]);
    atomicMax(&mEnc[(size_t)d * NH + hh], encf(x));
  }
}

// ---------------------------------------------------------------------------
// Kernel 4: ex = exp(e - m[dst]); segment-sum via hardware fp32 atomics
// ---------------------------------------------------------------------------
__global__ __launch_bounds__(256) void edge_exp_kernel(
    const int* __restrict__ src, const int* __restrict__ dst,
    const float* __restrict__ es, const float* __restrict__ ed,
    const unsigned* __restrict__ mEnc, float* __restrict__ ex,
    float* __restrict__ ssum, int E)
{
  int e = blockIdx.x * blockDim.x + threadIdx.x;
  if (e >= E) return;
  int s = src[e], d = dst[e];
#pragma unroll
  for (int hh = 0; hh < NH; ++hh) {
    float x = lrelu(es[(size_t)s * NH + hh] + ed[(size_t)d * NH + hh]);
    float m = decf(mEnc[(size_t)d * NH + hh]);
    float v = __expf(x - m);
    ex[(size_t)e * NH + hh] = v;
    unsafeAtomicAdd(&ssum[(size_t)d * NH + hh], v);
  }
}

// ---------------------------------------------------------------------------
// Kernel 5: out[dst] += h[src] * a  (one block of 256 threads per edge,
// each thread handles 2 of the 512 features; h[src] gathers hit L2)
// ---------------------------------------------------------------------------
__global__ __launch_bounds__(256) void agg_kernel(
    const int* __restrict__ src, const int* __restrict__ dst,
    const float* __restrict__ ex, const float* __restrict__ ssum,
    const float* __restrict__ dist, const float* __restrict__ h,
    float* __restrict__ out)
{
  __shared__ float a_sh[NH];
  int e = blockIdx.x;
  int s = src[e], d = dst[e];
  if (threadIdx.x < NH) {
    int hh = threadIdx.x;
    float denom = ssum[(size_t)d * NH + hh];
    a_sh[hh] = ex[(size_t)e * NH + hh] / denom * dist[(size_t)e * NH + hh];
  }
  __syncthreads();
  const float* hs = h + (size_t)s * HD;
  float* od = out + (size_t)d * HD;
#pragma unroll
  for (int rep = 0; rep < 2; ++rep) {
    int idx = threadIdx.x + rep * 256;
    float v = hs[idx] * a_sh[idx >> 6];
    unsafeAtomicAdd(&od[idx], v);
  }
}

// ---------------------------------------------------------------------------
extern "C" void kernel_launch(void* const* d_in, const int* in_sizes, int n_in,
                              void* d_out, int out_size, void* d_ws, size_t ws_size,
                              hipStream_t stream) {
  const float* feat = (const float*)d_in[0];
  const float* dist = (const float*)d_in[1];
  const int*   src  = (const int*)d_in[2];
  const int*   dst  = (const int*)d_in[3];
  const float* W    = (const float*)d_in[4];
  const float* bias = (const float*)d_in[5];
  const float* was  = (const float*)d_in[6];
  const float* wad  = (const float*)d_in[7];
  float* out = (float*)d_out;
  const int E = in_sizes[2];

  // workspace carve-up (~186 MB)
  char* ws = (char*)d_ws;
  float*     h     = (float*)ws;     ws += (size_t)NN * HD * sizeof(float);
  _Float16*  featH = (_Float16*)ws;  ws += (size_t)NN * IN_D * sizeof(_Float16);
  _Float16*  WH    = (_Float16*)ws;  ws += (size_t)HD * IN_D * sizeof(_Float16);
  float*     es    = (float*)ws;     ws += (size_t)NN * NH * sizeof(float);
  float*     ed    = (float*)ws;     ws += (size_t)NN * NH * sizeof(float);
  unsigned*  mEnc  = (unsigned*)ws;  ws += (size_t)NN * NH * sizeof(unsigned);
  float*     ssum  = (float*)ws;     ws += (size_t)NN * NH * sizeof(float);
  float*     ex    = (float*)ws;     // E * NH floats

  init_kernel<<<2048, 256, 0, stream>>>(out, ssum, mEnc);

  cvt_f16_kernel<<<2048, 256, 0, stream>>>(feat, W, featH, WH);

  dim3 ggrid(NN / 16, HD / 128);  // 3125 x 4 wave-tiles (16x128 each)
  gemm_lrelu_kernel<<<ggrid, 32, 0, stream>>>(featH, WH, bias, h);

  logits_kernel<<<(NN * NH + 255) / 256, 256, 0, stream>>>(h, was, wad, es, ed);

  edge_max_kernel<<<(E + 255) / 256, 256, 0, stream>>>(src, dst, es, ed, mEnc, E);

  edge_exp_kernel<<<(E + 255) / 256, 256, 0, stream>>>(src, dst, es, ed, mEnc, ex, ssum, E);

  agg_kernel<<<E, 256, 0, stream>>>(src, dst, ex, ssum, dist, h, out);
}
